// WeightedSumGraphRepresentation_26972394619596
// MI455X (gfx1250) — matrix operand order repaired and checked by reading, not compiled, not run on hardware
//
#include <hip/hip_runtime.h>
#include <math.h>

#define V_NODES 1000000
#define VD      256
#define GD      256
#define NH      16
#define NG      50000
#define HID     128
#define NTILES  (V_NODES / 16)   // 62500 node tiles of 16

typedef __attribute__((ext_vector_type(16))) __bf16        v16bf;
typedef __attribute__((ext_vector_type(8)))  float         v8f;
typedef __attribute__((ext_vector_type(4)))  float         f32x4;
typedef __attribute__((ext_vector_type(4)))  unsigned int  u32x4;
typedef __attribute__((ext_vector_type(8)))  unsigned int  u32x8;

union BFrag  { v16bf v; unsigned int d[8]; unsigned short u[16]; };
union F32x8  { v8f  v; float f[8]; };

// round-to-nearest bf16 (bias-add) packed pair via one v_perm_b32
__device__ __forceinline__ unsigned pack2bf(float a, float b) {
    unsigned ua = __float_as_uint(a) + 0x8000u;
    unsigned ub = __float_as_uint(b) + 0x8000u;
    return __builtin_amdgcn_perm(ub, ua, 0x07060302u);   // {hi16(b), hi16(a)}
}
__device__ __forceinline__ unsigned short f2bf(float x) {
    return (unsigned short)((__float_as_uint(x) + 0x8000u) >> 16);
}

// ---- Tensor Data Mover: copy 16x256 f32 tile (contiguous 4096 elems) -> LDS
__device__ __forceinline__ void tdm_load_tile_f32(const float* gsrc, unsigned lds_off) {
    unsigned long long ga = (unsigned long long)(uintptr_t)gsrc;
    unsigned ga_lo = __builtin_amdgcn_readfirstlane((unsigned)ga);
    unsigned ga_hi = __builtin_amdgcn_readfirstlane((unsigned)(ga >> 32));
    unsigned lo    = __builtin_amdgcn_readfirstlane(lds_off);

    u32x4 g0;
    g0[0] = 1u;                                   // count=1 (valid user D#)
    g0[1] = lo;                                   // lds_addr (bytes)
    g0[2] = ga_lo;                                // global_addr[31:0]
    g0[3] = (ga_hi & 0x01FFFFFFu) | (2u << 30);   // global_addr[56:32] | type=2

    u32x8 g1;
    g1[0] = 2u << 16;        // data_size = 4 bytes
    g1[1] = 4096u << 16;     // tensor_dim0 = 4096 (bits 79:48, low half)
    g1[2] = 1u << 16;        // tensor_dim1 = 1
    g1[3] = 4096u << 16;     // tile_dim0 = 4096
    g1[4] = 1u;              // tile_dim1 = 1
    g1[5] = 4096u;           // tensor_dim0_stride
    g1[6] = 0u;
    g1[7] = 0u;

    asm volatile("tensor_load_to_lds %0, %1" :: "s"(g0), "s"(g1) : "memory");
    __builtin_amdgcn_s_wait_tensorcnt(0);
}

// Build one 16x32 bf16 A fragment from the f32 LDS tile (row-major, stride RS)
__device__ __forceinline__ v16bf build_afrag(const float* rowp) {
    BFrag f;
    f32x4 q0 = *(const f32x4*)(rowp + 0);
    f32x4 q1 = *(const f32x4*)(rowp + 4);
    f32x4 q2 = *(const f32x4*)(rowp + 16);
    f32x4 q3 = *(const f32x4*)(rowp + 20);
    f.d[0] = pack2bf(q0[0], q0[1]); f.d[1] = pack2bf(q0[2], q0[3]);
    f.d[2] = pack2bf(q1[0], q1[1]); f.d[3] = pack2bf(q1[2], q1[3]);
    f.d[4] = pack2bf(q2[0], q2[1]); f.d[5] = pack2bf(q2[2], q2[3]);
    f.d[6] = pack2bf(q3[0], q3[1]); f.d[7] = pack2bf(q3[2], q3[3]);
    return f.v;
}

// Store relu'd hidden value at (row m, col nt*16+l15) in layer-2 A-frag order
__device__ __forceinline__ void store_hidden(unsigned short* hb, float h,
                                             int m, int nt, int l15) {
    int hh = (l15 >> 3) & 1;
    int e  = (l15 & 7) + (nt & 1) * 8;
    int l2 = hh * 16 + m;
    hb[((nt >> 1) * 32 + l2) * 16 + e] = f2bf(h);
}

// ---------------------------------------------------------------- utilities
__global__ void pack_weights_kernel(const float* __restrict__ w,
                                    unsigned short* __restrict__ o,
                                    int K, int N) {
    int i = blockIdx.x * 256 + threadIdx.x;
    if (i >= K * N) return;
    int e    = i & 15;
    int ln   = (i >> 4) & 31;
    int frag = i >> 9;
    int KC   = K >> 5;
    int kc   = frag % KC;
    int nt   = frag / KC;
    int k = 32 * kc + (e >> 3) * 16 + (ln >> 4) * 8 + (e & 7);
    int n = nt * 16 + (ln & 15);
    o[i] = f2bf(w[k * N + n]);
}

__global__ void init_buffers_kernel(float* __restrict__ seg_max,
                                    float* __restrict__ denom,
                                    float* __restrict__ out) {
    int i = blockIdx.x * 256 + threadIdx.x;
    if (i < NG * NH) { seg_max[i] = -3.0e38f; denom[i] = 0.0f; }
    if (i < NG * GD) { out[i] = 0.0f; }
}

// ----------------------------------------------------- scoring MLP (WMMA)
__global__ __launch_bounds__(128) void scoring_kernel(
    const float* __restrict__ x,
    const unsigned short* __restrict__ w1,   // packed [256,128]
    const unsigned short* __restrict__ w2,   // packed [128,16]
    float* __restrict__ scores)              // [V,16]
{
    __shared__ __align__(16) float lds[4 * 4096];    // 16 KB per wave
    const int lane = threadIdx.x & 31;
    const int wave = threadIdx.x >> 5;
    const int half = lane >> 4;
    const int l15  = lane & 15;
    float* myf = lds + wave * 4096;
    unsigned short* hb = (unsigned short*)myf;       // hidden frag region [0,4096) B

    int tile = blockIdx.x * 4 + wave;
    const bool active = (tile < NTILES);
    if (!active) tile = 0;
    const long base = (long)tile * 16;

    // DMA the whole X tile into LDS (wave-uniform descriptor), then wait
    tdm_load_tile_f32(x + base * VD, (unsigned)(uintptr_t)myf);

    // A fragments straight out of the f32 tile
    BFrag afrag[8];
    #pragma unroll
    for (int kc = 0; kc < 8; ++kc)
        afrag[kc].v = build_afrag(myf + l15 * VD + kc * 32 + half * 8);
    asm volatile("" ::: "memory");                   // frag reads before overwrite

    // layer 1: relu(X @ W1) -> hidden in fragment order; two tiles in flight
    #pragma unroll
    for (int nt = 0; nt < 8; nt += 2) {
        F32x8 acc0, acc1;
        #pragma unroll
        for (int i = 0; i < 8; ++i) { acc0.f[i] = 0.0f; acc1.f[i] = 0.0f; }
        #pragma unroll
        for (int kc = 0; kc < 8; ++kc) {
            BFrag b0, b1;
            b0.v = *(const v16bf*)(w1 + (((nt    ) * 8 + kc) * 32 + lane) * 16);
            b1.v = *(const v16bf*)(w1 + (((nt + 1) * 8 + kc) * 32 + lane) * 16);
            acc0.v = __builtin_amdgcn_wmma_f32_16x16x32_bf16(
                false, afrag[kc].v, false, b0.v, (short)0, acc0.v, false, false);
            acc1.v = __builtin_amdgcn_wmma_f32_16x16x32_bf16(
                false, afrag[kc].v, false, b1.v, (short)0, acc1.v, false, false);
        }
        #pragma unroll
        for (int i = 0; i < 8; ++i) {
            int m = i + half * 8;
            store_hidden(hb, fmaxf(acc0.f[i], 0.0f), m, nt,     l15);
            store_hidden(hb, fmaxf(acc1.f[i], 0.0f), m, nt + 1, l15);
        }
    }
    asm volatile("" ::: "memory");                   // hidden stores before reads

    // layer 2: scores[16x16] = hidden @ W2 (K=128 -> 4 WMMAs)
    F32x8 acc2;
    #pragma unroll
    for (int i = 0; i < 8; ++i) acc2.f[i] = 0.0f;
    #pragma unroll
    for (int kc = 0; kc < 4; ++kc) {
        BFrag a2, b2;
        a2.v = *(const v16bf*)(hb + kc * 512 + lane * 16);
        b2.v = *(const v16bf*)(w2 + (kc * 32 + lane) * 16);
        acc2.v = __builtin_amdgcn_wmma_f32_16x16x32_bf16(
            false, a2.v, false, b2.v, (short)0, acc2.v, false, false);
    }
    if (active) {
        #pragma unroll
        for (int i = 0; i < 8; ++i) {
            int m = i + half * 8;
            scores[(base + m) * NH + l15] = acc2.f[i];
        }
    }
}

// ---------------------------------------------- segment softmax (atomics)
__global__ void segmax_kernel(const float* __restrict__ scores,
                              const int* __restrict__ seg,
                              float* __restrict__ seg_max) {
    long i = (long)blockIdx.x * 256 + threadIdx.x;
    if (i >= (long)V_NODES * NH) return;
    int v = (int)(i >> 4), h = (int)(i & 15);
    float s = scores[i];
    unsigned int* addr = (unsigned int*)&seg_max[seg[v] * NH + h];
    unsigned int old = *addr;
    while (true) {
        float of = __uint_as_float(old);
        if (of >= s) break;
        unsigned int assumed = old;
        old = atomicCAS(addr, assumed, __float_as_uint(s));
        if (old == assumed) break;
    }
}

__global__ void softmax_num_kernel(float* __restrict__ scores,   // in-place -> exp
                                   const int* __restrict__ seg,
                                   const float* __restrict__ seg_max,
                                   float* __restrict__ denom) {
    long i = (long)blockIdx.x * 256 + threadIdx.x;
    if (i >= (long)V_NODES * NH) return;
    int v = (int)(i >> 4), h = (int)(i & 15);
    float e = __expf(scores[i] - seg_max[seg[v] * NH + h]);
    scores[i] = e;
    atomicAdd(&denom[seg[v] * NH + h], e);
}

// ------------------------- transform MLP + weighting + segment sum (WMMA)
__global__ __launch_bounds__(128) void transform_kernel(
    const float* __restrict__ x,
    const int* __restrict__ seg,
    const unsigned short* __restrict__ w1,   // packed [256,128]
    const unsigned short* __restrict__ w2,   // packed [128,256]
    const float* __restrict__ ex,            // [V,16] exp(score - max)
    const float* __restrict__ denom,         // [G,16]
    float* __restrict__ out)                 // [G,256]
{
    __shared__ __align__(16) float lds[4 * 4096];
    const int lane = threadIdx.x & 31;
    const int wave = threadIdx.x >> 5;
    const int half = lane >> 4;
    const int l15  = lane & 15;
    float* myf = lds + wave * 4096;
    unsigned short* hb = (unsigned short*)myf;       // hidden frags [0,4096) B
    float* wt = myf + 1024;                          // softmax weights [4096,5120) B

    int tile = blockIdx.x * 4 + wave;
    const bool active = (tile < NTILES);
    if (!active) tile = 0;
    const long base = (long)tile * 16;

    tdm_load_tile_f32(x + base * VD, (unsigned)(uintptr_t)myf);

    BFrag afrag[8];
    #pragma unroll
    for (int kc = 0; kc < 8; ++kc)
        afrag[kc].v = build_afrag(myf + l15 * VD + kc * 32 + half * 8);
    asm volatile("" ::: "memory");

    // softmax weights for this tile: wt[node][head] = ex / denom[seg]
    #pragma unroll
    for (int i = 0; i < 8; ++i) {
        int idx  = i * 32 + lane;                    // 0..255
        int node = idx >> 4, h = idx & 15;
        long vi  = base + node;
        int  g   = seg[vi];
        wt[idx]  = ex[vi * NH + h] / denom[g * NH + h];
    }

    // layer 1: hidden = relu(X @ W1t) -> fragment-order LDS
    #pragma unroll
    for (int nt = 0; nt < 8; nt += 2) {
        F32x8 acc0, acc1;
        #pragma unroll
        for (int i = 0; i < 8; ++i) { acc0.f[i] = 0.0f; acc1.f[i] = 0.0f; }
        #pragma unroll
        for (int kc = 0; kc < 8; ++kc) {
            BFrag b0, b1;
            b0.v = *(const v16bf*)(w1 + (((nt    ) * 8 + kc) * 32 + lane) * 16);
            b1.v = *(const v16bf*)(w1 + (((nt + 1) * 8 + kc) * 32 + lane) * 16);
            acc0.v = __builtin_amdgcn_wmma_f32_16x16x32_bf16(
                false, afrag[kc].v, false, b0.v, (short)0, acc0.v, false, false);
            acc1.v = __builtin_amdgcn_wmma_f32_16x16x32_bf16(
                false, afrag[kc].v, false, b1.v, (short)0, acc1.v, false, false);
        }
        #pragma unroll
        for (int i = 0; i < 8; ++i) {
            int m = i + half * 8;
            store_hidden(hb, fmaxf(acc0.f[i], 0.0f), m, nt,     l15);
            store_hidden(hb, fmaxf(acc1.f[i], 0.0f), m, nt + 1, l15);
        }
    }
    asm volatile("" ::: "memory");

    // graph ids for my 8 rows
    int gidx[8];
    #pragma unroll
    for (int i = 0; i < 8; ++i) gidx[i] = seg[base + i + half * 8];

    // layer-2 A fragments (K = 128)
    BFrag a2[4];
    #pragma unroll
    for (int kc = 0; kc < 4; ++kc)
        a2[kc].v = *(const v16bf*)(hb + kc * 512 + lane * 16);

    // layer 2: 16 N-tiles (each == one head), two in flight; weight + scatter
    #pragma unroll 2
    for (int nt = 0; nt < 16; nt += 2) {
        F32x8 acc0, acc1;
        #pragma unroll
        for (int i = 0; i < 8; ++i) { acc0.f[i] = 0.0f; acc1.f[i] = 0.0f; }
        #pragma unroll
        for (int kc = 0; kc < 4; ++kc) {
            BFrag b0, b1;
            b0.v = *(const v16bf*)(w2 + (((nt    ) * 4 + kc) * 32 + lane) * 16);
            b1.v = *(const v16bf*)(w2 + (((nt + 1) * 4 + kc) * 32 + lane) * 16);
            acc0.v = __builtin_amdgcn_wmma_f32_16x16x32_bf16(
                false, a2[kc].v, false, b0.v, (short)0, acc0.v, false, false);
            acc1.v = __builtin_amdgcn_wmma_f32_16x16x32_bf16(
                false, a2[kc].v, false, b1.v, (short)0, acc1.v, false, false);
        }
        if (active) {
            #pragma unroll
            for (int i = 0; i < 8; ++i) {
                int m = i + half * 8;
                float r0 = fmaxf(acc0.f[i], 0.0f);   // outer relu
                float r1 = fmaxf(acc1.f[i], 0.0f);
                atomicAdd(&out[(long)gidx[i] * GD + (nt    ) * 16 + l15],
                          r0 * wt[m * 16 + nt]);
                atomicAdd(&out[(long)gidx[i] * GD + (nt + 1) * 16 + l15],
                          r1 * wt[m * 16 + nt + 1]);
            }
        }
    }
}

// ------------------------------------------------------------------ launch
extern "C" void kernel_launch(void* const* d_in, const int* in_sizes, int n_in,
                              void* d_out, int out_size, void* d_ws, size_t ws_size,
                              hipStream_t stream) {
    (void)in_sizes; (void)n_in; (void)out_size; (void)ws_size;

    const float* x   = (const float*)d_in[0];
    const int*   seg = (const int*)d_in[1];
    // d_in[2] = num_graphs (compile-time constant NG)
    const float* sw1 = (const float*)d_in[3];
    const float* sw2 = (const float*)d_in[4];
    const float* tw1 = (const float*)d_in[5];
    const float* tw2 = (const float*)d_in[6];
    float* out = (float*)d_out;

    char* ws = (char*)d_ws;
    float*          scores = (float*)(ws);                     // V*16*4  = 64,000,000
    float*          segmax = (float*)(ws + 64000000);          // G*16*4  =  3,200,000
    float*          denom  = (float*)(ws + 67200000);          // G*16*4  =  3,200,000
    unsigned short* w1s    = (unsigned short*)(ws + 70400000); // 65,536
    unsigned short* w2s    = (unsigned short*)(ws + 70465536); //  4,096
    unsigned short* w1t    = (unsigned short*)(ws + 70469632); // 65,536
    unsigned short* w2t    = (unsigned short*)(ws + 70535168); // 65,536

    pack_weights_kernel<<<(VD * HID + 255) / 256, 256, 0, stream>>>(sw1, w1s, VD, HID);
    pack_weights_kernel<<<(HID * NH + 255) / 256, 256, 0, stream>>>(sw2, w2s, HID, NH);
    pack_weights_kernel<<<(VD * HID + 255) / 256, 256, 0, stream>>>(tw1, w1t, VD, HID);
    pack_weights_kernel<<<(HID * GD + 255) / 256, 256, 0, stream>>>(tw2, w2t, HID, GD);

    init_buffers_kernel<<<(NG * GD + 255) / 256, 256, 0, stream>>>(segmax, denom, out);

    scoring_kernel<<<(NTILES + 3) / 4, 128, 0, stream>>>(x, w1s, w2s, scores);

    long vh = (long)V_NODES * NH;
    segmax_kernel<<<(int)((vh + 255) / 256), 256, 0, stream>>>(scores, seg, segmax);
    softmax_num_kernel<<<(int)((vh + 255) / 256), 256, 0, stream>>>(scores, seg, segmax, denom);

    transform_kernel<<<(NTILES + 3) / 4, 128, 0, stream>>>(x, seg, w1t, w2t,
                                                           scores, denom, out);
}
